// LArVoxelClassifier_48593259987034
// MI455X (gfx1250) — matrix operand order repaired and checked by reading, not compiled, not run on hardware
//
#include <hip/hip_runtime.h>

#define DEV_INLINE __device__ __forceinline__

#if __has_builtin(__builtin_amdgcn_global_load_async_to_lds_b128) && \
    __has_builtin(__builtin_amdgcn_s_wait_asynccnt)
#define HAVE_ASYNC_LDS 1
#else
#define HAVE_ASYNC_LDS 0
#endif

namespace {
enum : int { N0_ = 100000, N1_ = 50000, N2_ = 25000, N3_ = 12500, N4_ = 6250 };
constexpr int NB = 4;      // batch instances
constexpr int KK = 27;     // rulebook offsets
constexpr float EPS_C = 1e-5f;
constexpr float SLOPE_C = 0.01f;
// monotone-mapped -inf (= ~bits(-inf)); init value for float atomic-max-as-uint
constexpr unsigned FMAP_NEG_INF = 0x007FFFFFu;
}

typedef __attribute__((ext_vector_type(16))) __bf16          v16bf;
typedef __attribute__((ext_vector_type(8)))  float           v8f;
typedef __attribute__((ext_vector_type(16))) unsigned short  u16x16;
typedef __attribute__((ext_vector_type(8)))  unsigned short  u16x8;
typedef int vsi4 __attribute__((vector_size(16)));  // int4 pointee for async-lds builtin

#if HAVE_ASYNC_LDS
typedef __attribute__((address_space(1))) vsi4* g4p_t;  // global int4*
typedef __attribute__((address_space(3))) vsi4* l4p_t;  // LDS int4*
#endif

union BFPack { u16x16 u; v16bf b; };

DEV_INLINE unsigned short f2bf(float f) {
  unsigned u = __float_as_uint(f);
  u += 0x7FFFu + ((u >> 16) & 1u);   // round-to-nearest-even bf16
  return (unsigned short)(u >> 16);
}

// monotone float<->uint mapping for atomic max on floats (incl. negatives)
DEV_INLINE unsigned fmap(float f) {
  unsigned u = __float_as_uint(f);
  return (u & 0x80000000u) ? ~u : (u | 0x80000000u);
}
DEV_INLINE float funmap(unsigned u) {
  u = (u & 0x80000000u) ? (u ^ 0x80000000u) : ~u;
  return __uint_as_float(u);
}

// ---------------------------------------------------------------------------
// Weight convert: W [27, CIN, COUT] f32  ->  wbt [27, COUT, CINP] bf16,
// transposed (K contiguous per output column) and zero-padded to CINP.
// ---------------------------------------------------------------------------
template <int CIN, int COUT>
__global__ void wconv_kernel(const float* __restrict__ W,
                             unsigned short* __restrict__ wbt) {
  constexpr int CINP = (CIN < 32) ? 32 : CIN;
  int idx = blockIdx.x * blockDim.x + threadIdx.x;
  if (idx >= KK * COUT * CINP) return;
  int c = idx % CINP;
  int n = (idx / CINP) % COUT;
  int k = idx / (CINP * COUT);
  float v = (c < CIN) ? W[((size_t)k * CIN + c) * COUT + n] : 0.0f;
  wbt[idx] = f2bf(v);
}

// ---------------------------------------------------------------------------
// Rulebook sparse conv: out[i,:] = sum_k x[nbr[i,k],:] @ W[k]
// x is bf16 (halves gather HBM traffic vs f32). One block = 32 output rows
// (2 M-tiles); 8 waves split the Cout/16 N-tiles; each B fragment feeds 2
// WMMAs. Gather goes global->LDS via async-to-LDS (ASYNCcnt) when available.
// f32 accumulate via v_wmma_f32_16x16x32_bf16.
// ---------------------------------------------------------------------------
template <int CIN, int COUT>
__global__ __launch_bounds__(256) void spconv_wmma_kernel(
    const unsigned short* __restrict__ x, const int* __restrict__ nbr,
    const unsigned short* __restrict__ wbt, float* __restrict__ out,
    int n_out) {
  constexpr int CINP = (CIN < 32) ? 32 : CIN;
  constexpr int MT = 2;                  // M tiles per block
  constexpr int ROWS = MT * 16;
  constexpr int NT = COUT / 16;          // N tiles
  constexpr int TPW = (NT + 7) / 8;      // N tiles per wave (8 waves)
  __shared__ unsigned short sA[ROWS * CINP];

  const int base = blockIdx.x * ROWS;
  const int tid = (int)threadIdx.x;
  const int lane = tid & 31;
  const int wv = tid >> 5;
  const int ln = lane & 15;
  const bool hi = lane >= 16;

  v8f acc[TPW][MT];
#pragma unroll
  for (int t = 0; t < TPW; ++t)
#pragma unroll
    for (int mt = 0; mt < MT; ++mt)
#pragma unroll
      for (int j = 0; j < 8; ++j) acc[t][mt][j] = 0.0f;

  for (int k = 0; k < KK; ++k) {
    __syncthreads();
    if constexpr (CIN >= 32) {
      // 16-byte-segment gather of ROWS bf16 rows into LDS
      constexpr int SEGR = CINP / 8;     // 16B segments per row
      constexpr int SEGS = ROWS * SEGR;
      for (int s = tid; s < SEGS; s += 256) {
        int r = s / SEGR, ss = s % SEGR;
        int i = base + r;
        if (i >= n_out) i = n_out - 1;
        int src = nbr[i * KK + k];
        const unsigned short* gp = x + (size_t)src * CIN + ss * 8;
        unsigned short* lp = &sA[r * CINP + ss * 8];
#if HAVE_ASYNC_LDS
        __builtin_amdgcn_global_load_async_to_lds_b128(
            (g4p_t)(unsigned long long)gp,
            (l4p_t)(unsigned)(unsigned long long)lp, 0, 0);
#else
        *(u16x8*)lp = *(const u16x8*)gp;
#endif
      }
#if HAVE_ASYNC_LDS
      __builtin_amdgcn_s_wait_asynccnt(0);
#endif
    } else {
      // narrow input (CIN=3): scalar gather with zero padding to CINP
      for (int e = tid; e < ROWS * CINP; e += 256) {
        int r = e / CINP, c = e % CINP;
        int i = base + r;
        if (i >= n_out) i = n_out - 1;
        unsigned short v = 0;
        if (c < CIN) v = x[(size_t)nbr[i * KK + k] * CIN + c];
        sA[e] = v;
      }
    }
    if (tid < ROWS && k + 1 < KK) {  // prefetch next rulebook offset's rows
      int i = base + tid;
      if (i >= n_out) i = n_out - 1;
      int src = nbr[i * KK + k + 1];
      __builtin_prefetch(&x[(size_t)src * CIN], 0, 1);
    }
    __syncthreads();

    for (int chunk = 0; chunk < CINP; chunk += 32) {
      // A fragments: lane<16 holds row, K {0..7,16..23}; lane>=16 {8..15,24..31}
      BFPack ua[MT];
#pragma unroll
      for (int mt = 0; mt < MT; ++mt) {
        const unsigned short* pa =
            &sA[(mt * 16 + ln) * CINP + chunk + (hi ? 8 : 0)];
        u16x8 a0 = *(const u16x8*)pa;
        u16x8 a1 = *(const u16x8*)(pa + 16);
        ua[mt].u = __builtin_shufflevector(a0, a1, 0, 1, 2, 3, 4, 5, 6, 7, 8,
                                           9, 10, 11, 12, 13, 14, 15);
      }
#pragma unroll
      for (int t = 0; t < TPW; ++t) {
        int nt = wv + t * 8;
        if (nt < NT) {
          // B fragment: column = ln; lane<16 K {0..7,16..23}, lane>=16 K {8..15,24..31}
          const unsigned short* pb =
              wbt + ((size_t)(k * COUT + nt * 16 + ln)) * CINP + chunk +
              (hi ? 8 : 0);
          BFPack ub;
          ub.u = __builtin_shufflevector(*(const u16x8*)pb,
                                         *(const u16x8*)(pb + 16), 0, 1, 2, 3,
                                         4, 5, 6, 7, 8, 9, 10, 11, 12, 13, 14,
                                         15);
#pragma unroll
          for (int mt = 0; mt < MT; ++mt) {
            acc[t][mt] = __builtin_amdgcn_wmma_f32_16x16x32_bf16(
                false, ua[mt].b, false, ub.b, (short)0, acc[t][mt], false,
                false);
          }
        }
      }
    }
  }

  // D layout: lanes 0-15 VGPR j -> M=j; lanes 16-31 VGPR j -> M=j+8; N=ln
  const int mbase = hi ? 8 : 0;
#pragma unroll
  for (int t = 0; t < TPW; ++t) {
    int nt = wv + t * 8;
    if (nt < NT) {
#pragma unroll
      for (int mt = 0; mt < MT; ++mt) {
#pragma unroll
        for (int j = 0; j < 8; ++j) {
          int row = base + mt * 16 + mbase + j;
          if (row < n_out)
            out[(size_t)row * COUT + nt * 16 + ln] = acc[t][mt][j];
        }
      }
    }
  }
}

// ---------------------------------------------------------------------------
// Utility / normalization / pooling / head kernels
// ---------------------------------------------------------------------------
__global__ void fillf_kernel(float* __restrict__ p, float v, int n) {
  int i = blockIdx.x * blockDim.x + threadIdx.x;
  if (i < n) p[i] = v;
}
__global__ void fillu_kernel(unsigned* __restrict__ p, unsigned v, int n) {
  int i = blockIdx.x * blockDim.x + threadIdx.x;
  if (i < n) p[i] = v;
}
__global__ void f2bf_kernel(const float* __restrict__ in,
                            unsigned short* __restrict__ outb, int n) {
  int i = blockIdx.x * blockDim.x + threadIdx.x;
  if (i < n) outb[i] = f2bf(in[i]);
}

__global__ void seg_stats_kernel(const float* __restrict__ x,
                                 const int* __restrict__ bidx, int N, int C,
                                 float* __restrict__ sums,
                                 float* __restrict__ sumsq,
                                 float* __restrict__ cnt) {
  int idx = blockIdx.x * blockDim.x + threadIdx.x;
  if (idx >= N * C) return;
  int n = idx / C, c = idx % C;
  int b = bidx ? bidx[n] : n;
  float v = x[idx];
  atomicAdd(&sums[b * C + c], v);
  atomicAdd(&sumsq[b * C + c], v * v);
  if (c == 0) atomicAdd(&cnt[b], 1.0f);
}

// in-place instance-norm + LeakyReLU; optionally writes a bf16 copy that the
// next sparse conv gathers from (halves its HBM gather traffic)
__global__ void inorm_lrelu_kernel(float* __restrict__ x,
                                   unsigned short* __restrict__ xb,
                                   const int* __restrict__ bidx, int N, int C,
                                   const float* __restrict__ sums,
                                   const float* __restrict__ sumsq,
                                   const float* __restrict__ cnt) {
  int idx = blockIdx.x * blockDim.x + threadIdx.x;
  if (idx >= N * C) return;
  int n = idx / C, c = idx % C;
  int b = bidx ? bidx[n] : n;
  float ct = fmaxf(cnt[b], 1.0f);
  float mean = sums[b * C + c] / ct;
  float var = sumsq[b * C + c] / ct - mean * mean;
  float v = (x[idx] - mean) * rsqrtf(var + EPS_C);
  v = (v >= 0.0f) ? v : SLOPE_C * v;
  x[idx] = v;
  if (xb) xb[idx] = f2bf(v);
}

// concat multiscale features directly into bf16 (only consumer is a conv)
__global__ void concat_kernel(const float* __restrict__ h0,
                              const float* __restrict__ h1,
                              const float* __restrict__ h2,
                              const float* __restrict__ h3,
                              const float* __restrict__ h4,
                              const int* __restrict__ up1,
                              const int* __restrict__ up2,
                              const int* __restrict__ up3,
                              const int* __restrict__ up4,
                              unsigned short* __restrict__ y) {
  long long idx = (long long)blockIdx.x * blockDim.x + threadIdx.x;
  if (idx >= (long long)N0_ * 512) return;
  int i = (int)(idx / 512), c = (int)(idx % 512);
  float v;
  if (c < 32)       v = h0[(size_t)i * 32 + c];
  else if (c < 64)  v = h1[(size_t)up1[i] * 32 + (c - 32)];
  else if (c < 128) v = h2[(size_t)up2[i] * 64 + (c - 64)];
  else if (c < 256) v = h3[(size_t)up3[i] * 128 + (c - 128)];
  else              v = h4[(size_t)up4[i] * 256 + (c - 256)];
  y[idx] = f2bf(v);
}

__global__ void pool_kernel(const float* __restrict__ m3,
                            const int* __restrict__ bidx, int N,
                            unsigned* __restrict__ gmaxu,
                            float* __restrict__ gsum,
                            float* __restrict__ cnt) {
  int idx = blockIdx.x * blockDim.x + threadIdx.x;
  if (idx >= N * 512) return;
  int n = idx / 512, c = idx % 512;
  int b = bidx[n];
  float v = m3[idx];
  atomicMax(&gmaxu[b * 512 + c], fmap(v));
  atomicAdd(&gsum[b * 512 + c], v);
  if (c == 0) atomicAdd(&cnt[b], 1.0f);
}

__global__ void gbuild_kernel(const unsigned* __restrict__ gmaxu,
                              const float* __restrict__ gsum,
                              const float* __restrict__ cnt,
                              float* __restrict__ g) {
  int idx = blockIdx.x * blockDim.x + threadIdx.x;
  if (idx >= NB * 1024) return;
  int b = idx / 1024, c = idx % 1024;
  g[idx] = (c < 512) ? funmap(gmaxu[b * 512 + c])
                     : gsum[b * 512 + (c - 512)] / fmaxf(cnt[b], 1.0f);
}

__global__ void linear_kernel(const float* __restrict__ x,
                              const float* __restrict__ W,
                              const float* __restrict__ bias,
                              float* __restrict__ y, int M, int Kd, int N) {
  int idx = blockIdx.x * blockDim.x + threadIdx.x;
  if (idx >= M * N) return;
  int m = idx / N, n = idx % N;
  float acc = bias ? bias[n] : 0.0f;
  for (int k2 = 0; k2 < Kd; ++k2) acc += x[m * Kd + k2] * W[k2 * N + n];
  y[idx] = acc;
}

// ---------------------------------------------------------------------------
// Host-side layer driver
// ---------------------------------------------------------------------------
template <int CIN, int COUT>
static void run_conv(const unsigned short* xb, const int* nbr, const float* W,
                     const int* bidx, unsigned short* wbt, float* out,
                     unsigned short* outb, int n_out, float* sums,
                     float* sumsq, float* cnt, hipStream_t stream) {
  constexpr int CINP = (CIN < 32) ? 32 : CIN;
  int wtot = KK * COUT * CINP;
  wconv_kernel<CIN, COUT><<<(wtot + 255) / 256, 256, 0, stream>>>(W, wbt);
  spconv_wmma_kernel<CIN, COUT>
      <<<(n_out + 31) / 32, 256, 0, stream>>>(xb, nbr, wbt, out, n_out);
  int statsN = 2 * NB * 512 + NB;  // sums|sumsq|cnt contiguous
  fillf_kernel<<<(statsN + 255) / 256, 256, 0, stream>>>(sums, 0.0f, statsN);
  int tot = n_out * COUT;
  seg_stats_kernel<<<(tot + 255) / 256, 256, 0, stream>>>(out, bidx, n_out,
                                                          COUT, sums, sumsq, cnt);
  inorm_lrelu_kernel<<<(tot + 255) / 256, 256, 0, stream>>>(
      out, outb, bidx, n_out, COUT, sums, sumsq, cnt);
}

extern "C" void kernel_launch(void* const* d_in, const int* in_sizes, int n_in,
                              void* d_out, int out_size, void* d_ws,
                              size_t ws_size, hipStream_t stream) {
  (void)in_sizes; (void)n_in; (void)out_size; (void)ws_size;
  const float* feats  = (const float*)d_in[0];
  const int* bidx0    = (const int*)d_in[1];
  const int* nbr0     = (const int*)d_in[2];
  const int* nbr1     = (const int*)d_in[3];
  const int* bidx1    = (const int*)d_in[4];
  const int* nbr2     = (const int*)d_in[5];
  const int* bidx2    = (const int*)d_in[6];
  const int* nbr3     = (const int*)d_in[7];
  const int* bidx3    = (const int*)d_in[8];
  const int* nbr4     = (const int*)d_in[9];
  const int* bidx4    = (const int*)d_in[10];
  const int* up1      = (const int*)d_in[11];
  const int* up2      = (const int*)d_in[12];
  const int* up3      = (const int*)d_in[13];
  const int* up4      = (const int*)d_in[14];
  const int* nbr_m1   = (const int*)d_in[15];
  const int* bidx_m1  = (const int*)d_in[16];
  const int* nbr_m2   = (const int*)d_in[17];
  const int* bidx_m2  = (const int*)d_in[18];
  const int* nbr_m3   = (const int*)d_in[19];
  const int* bidx_m3  = (const int*)d_in[20];
  const float* W0     = (const float*)d_in[21];
  const float* W1     = (const float*)d_in[22];
  const float* W2     = (const float*)d_in[23];
  const float* W3     = (const float*)d_in[24];
  const float* W4     = (const float*)d_in[25];
  const float* Wm1    = (const float*)d_in[26];
  const float* Wm2    = (const float*)d_in[27];
  const float* Wm3    = (const float*)d_in[28];
  const float* L1     = (const float*)d_in[29];
  const float* L2     = (const float*)d_in[30];
  const float* L3     = (const float*)d_in[31];
  const float* b3     = (const float*)d_in[32];

  char* p = (char*)d_ws;
  auto alloc = [&](size_t bytes) -> void* {
    void* r = (void*)p;
    p += (bytes + 255) & ~(size_t)255;
    return r;
  };
  // f32 activations (normalized in place; m3 feeds pooling)
  float* h0 = (float*)alloc(sizeof(float) * (size_t)N0_ * 32);
  float* h1 = (float*)alloc(sizeof(float) * (size_t)N1_ * 32);
  float* h2 = (float*)alloc(sizeof(float) * (size_t)N2_ * 64);
  float* h3 = (float*)alloc(sizeof(float) * (size_t)N3_ * 128);
  float* h4 = (float*)alloc(sizeof(float) * (size_t)N4_ * 256);
  float* m1 = (float*)alloc(sizeof(float) * (size_t)N1_ * 128);
  float* m2 = (float*)alloc(sizeof(float) * (size_t)N2_ * 256);
  float* m3 = (float*)alloc(sizeof(float) * (size_t)N3_ * 512);
  // bf16 copies of everything a sparse conv gathers from
  unsigned short* featsb = (unsigned short*)alloc(sizeof(unsigned short) * (size_t)N0_ * 3);
  unsigned short* h0b = (unsigned short*)alloc(sizeof(unsigned short) * (size_t)N0_ * 32);
  unsigned short* h1b = (unsigned short*)alloc(sizeof(unsigned short) * (size_t)N1_ * 32);
  unsigned short* h2b = (unsigned short*)alloc(sizeof(unsigned short) * (size_t)N2_ * 64);
  unsigned short* h3b = (unsigned short*)alloc(sizeof(unsigned short) * (size_t)N3_ * 128);
  unsigned short* yb  = (unsigned short*)alloc(sizeof(unsigned short) * (size_t)N0_ * 512);
  unsigned short* m1b = (unsigned short*)alloc(sizeof(unsigned short) * (size_t)N1_ * 128);
  unsigned short* m2b = (unsigned short*)alloc(sizeof(unsigned short) * (size_t)N2_ * 256);
  // instance-norm stats (sums|sumsq|cnt contiguous)
  float* stats = (float*)alloc(sizeof(float) * (2 * NB * 512 + NB));
  float* sums = stats;
  float* sumsq = stats + NB * 512;
  float* cnt = stats + 2 * NB * 512;
  unsigned* gmaxu = (unsigned*)alloc(sizeof(unsigned) * NB * 512);
  float* gsum = (float*)alloc(sizeof(float) * NB * 512);
  float* cntp = (float*)alloc(sizeof(float) * NB);
  float* g  = (float*)alloc(sizeof(float) * NB * 1024);
  float* z1 = (float*)alloc(sizeof(float) * NB * 512);
  float* z2 = (float*)alloc(sizeof(float) * NB * 512);
  // transposed padded bf16 weights
  unsigned short* wbt0  = (unsigned short*)alloc(sizeof(unsigned short) * KK * 32 * 32);
  unsigned short* wbt1  = (unsigned short*)alloc(sizeof(unsigned short) * KK * 32 * 32);
  unsigned short* wbt2  = (unsigned short*)alloc(sizeof(unsigned short) * KK * 64 * 32);
  unsigned short* wbt3  = (unsigned short*)alloc(sizeof(unsigned short) * KK * 128 * 64);
  unsigned short* wbt4  = (unsigned short*)alloc(sizeof(unsigned short) * KK * 256 * 128);
  unsigned short* wbtm1 = (unsigned short*)alloc(sizeof(unsigned short) * KK * 128 * 512);
  unsigned short* wbtm2 = (unsigned short*)alloc(sizeof(unsigned short) * KK * 256 * 128);
  unsigned short* wbtm3 = (unsigned short*)alloc(sizeof(unsigned short) * KK * 512 * 256);

  // ---- encoder ----
  f2bf_kernel<<<(N0_ * 3 + 255) / 256, 256, 0, stream>>>(feats, featsb, N0_ * 3);
  run_conv<3, 32>(featsb, nbr0, W0, bidx0, wbt0, h0, h0b, N0_, sums, sumsq, cnt, stream);
  run_conv<32, 32>(h0b, nbr1, W1, bidx1, wbt1, h1, h1b, N1_, sums, sumsq, cnt, stream);
  run_conv<32, 64>(h1b, nbr2, W2, bidx2, wbt2, h2, h2b, N2_, sums, sumsq, cnt, stream);
  run_conv<64, 128>(h2b, nbr3, W3, bidx3, wbt3, h3, h3b, N3_, sums, sumsq, cnt, stream);
  run_conv<128, 256>(h3b, nbr4, W4, bidx4, wbt4, h4, nullptr, N4_, sums, sumsq, cnt, stream);

  // ---- concat multiscale features -> yb [N0, 512] bf16 ----
  long long ytot = (long long)N0_ * 512;
  concat_kernel<<<(int)((ytot + 255) / 256), 256, 0, stream>>>(
      h0, h1, h2, h3, h4, up1, up2, up3, up4, yb);

  // ---- multiscale strided convs ----
  run_conv<512, 128>(yb, nbr_m1, Wm1, bidx_m1, wbtm1, m1, m1b, N1_, sums, sumsq, cnt, stream);
  run_conv<128, 256>(m1b, nbr_m2, Wm2, bidx_m2, wbtm2, m2, m2b, N2_, sums, sumsq, cnt, stream);
  run_conv<256, 512>(m2b, nbr_m3, Wm3, bidx_m3, wbtm3, m3, nullptr, N3_, sums, sumsq, cnt, stream);

  // ---- global max + avg pool ----
  fillu_kernel<<<(NB * 512 + 255) / 256, 256, 0, stream>>>(gmaxu, FMAP_NEG_INF, NB * 512);
  fillf_kernel<<<(NB * 512 + 255) / 256, 256, 0, stream>>>(gsum, 0.0f, NB * 512);
  fillf_kernel<<<1, 32, 0, stream>>>(cntp, 0.0f, NB);
  int ptot = N3_ * 512;
  pool_kernel<<<(ptot + 255) / 256, 256, 0, stream>>>(m3, bidx_m3, N3_, gmaxu, gsum, cntp);
  gbuild_kernel<<<(NB * 1024 + 255) / 256, 256, 0, stream>>>(gmaxu, gsum, cntp, g);

  // ---- MLP head (instance norm with one point per instance) ----
  int statsN = 2 * NB * 512 + NB;
  linear_kernel<<<(NB * 512 + 255) / 256, 256, 0, stream>>>(g, L1, (const float*)nullptr, z1, NB, 1024, 512);
  fillf_kernel<<<(statsN + 255) / 256, 256, 0, stream>>>(sums, 0.0f, statsN);
  seg_stats_kernel<<<(NB * 512 + 255) / 256, 256, 0, stream>>>(z1, nullptr, NB, 512, sums, sumsq, cnt);
  inorm_lrelu_kernel<<<(NB * 512 + 255) / 256, 256, 0, stream>>>(z1, nullptr, nullptr, NB, 512, sums, sumsq, cnt);

  linear_kernel<<<(NB * 512 + 255) / 256, 256, 0, stream>>>(z1, L2, (const float*)nullptr, z2, NB, 512, 512);
  fillf_kernel<<<(statsN + 255) / 256, 256, 0, stream>>>(sums, 0.0f, statsN);
  seg_stats_kernel<<<(NB * 512 + 255) / 256, 256, 0, stream>>>(z2, nullptr, NB, 512, sums, sumsq, cnt);
  inorm_lrelu_kernel<<<(NB * 512 + 255) / 256, 256, 0, stream>>>(z2, nullptr, nullptr, NB, 512, sums, sumsq, cnt);

  linear_kernel<<<1, 32, 0, stream>>>(z2, L3, b3, (float*)d_out, NB, 512, 6);
}
